// MultiHeadAttention_9586367005065
// MI455X (gfx1250) — matrix operand order repaired and checked by reading, not compiled, not run on hardware
//
#include <hip/hip_runtime.h>
#include <hip/hip_bf16.h>

// Problem constants
constexpr int SEQ    = 2048;
constexpr int DMODEL = 1024;
constexpr int NHEADS = 16;
constexpr int DKH    = 64;
constexpr int BATCH  = 2;
constexpr int MROWS  = BATCH * SEQ;          // 4096 rows for the projections

typedef __attribute__((ext_vector_type(16))) __bf16 v16bf;
typedef __attribute__((ext_vector_type(8)))  __bf16 v8bf;
typedef __attribute__((ext_vector_type(8)))  float  v8f;

union Frag16 { v16bf v; v8bf h[2]; };

__device__ __forceinline__ unsigned short f2bf(float f) {
    union { float f; unsigned u; } v; v.f = f;
    unsigned r = v.u + 0x7FFFu + ((v.u >> 16) & 1u);   // round-to-nearest-even
    return (unsigned short)(r >> 16);
}

__device__ __forceinline__ v8f wmma_bf16(const Frag16& a, const Frag16& b, v8f c) {
    return __builtin_amdgcn_wmma_f32_16x16x32_bf16(
        /*neg_a=*/false, a.v, /*neg_b=*/false, b.v,
        /*c_mod=*/(short)0, c, /*reuse_a=*/false, /*reuse_b=*/false);
}

// ---------------------------------------------------------------------------
// GEMM: C[4096,1024] = A[4096,1024] @ W[1024,1024] + bias
//   A_BF16:   A is bf16 (ushort) else fp32, converted to bf16 when staged.
//   OUT_MODE: 0 = fp32 row-major [M,N]                  (final out, NT store)
//             1 = bf16 per-head row-major [B,H,S,64]    (Q, K)
//             2 = bf16 per-head TRANSPOSED [B,H,64,S]   (V, for P@V B-frags)
// Block: 256 thr (8 waves), tile 128x128, K-step 32. Waves 4(M)x2(N),
// each wave 32x64 => 2x4 WMMA accumulators.
// ---------------------------------------------------------------------------
template<bool A_BF16, int OUT_MODE>
__global__ __launch_bounds__(256) void gemm_bias_k(const void* __restrict__ Aptr,
                                                   const float* __restrict__ W,
                                                   const float* __restrict__ bias,
                                                   void* __restrict__ outPtr) {
    __shared__ __align__(16) unsigned short ldsA[128][40];   // [m][k], pad 32->40
    __shared__ __align__(16) unsigned short ldsB[128][40];   // [n][k] (transposed)

    const int tid  = threadIdx.x;
    const int lane = tid & 31;
    const int wave = tid >> 5;
    const int bx   = blockIdx.x & 7;    // N tile (1024/128)
    const int by   = blockIdx.x >> 3;   // M tile (4096/128)
    const int m0   = by * 128;
    const int n0   = bx * 128;
    const int wm   = wave & 3;
    const int wn   = wave >> 2;
    const int hlf  = lane >> 4;

    const float*          Af = (const float*)Aptr;
    const unsigned short* Ab = (const unsigned short*)Aptr;

    const v8f vzero = {};
    v8f acc[2][4];
    #pragma unroll
    for (int i = 0; i < 2; ++i)
        #pragma unroll
        for (int j = 0; j < 4; ++j) acc[i][j] = vzero;

    for (int kb = 0; kb < DMODEL; kb += 32) {
        // --- stage A tile 128x32 (convert fp32->bf16 if needed) ---
        #pragma unroll
        for (int p = 0; p < 4; ++p) {
            int idx = p * 1024 + tid * 4;
            int r = idx >> 5, c = idx & 31;
            if (A_BF16) {
                ushort4 a4 = *(const ushort4*)&Ab[(m0 + r) * DMODEL + kb + c];
                *(ushort4*)&ldsA[r][c] = a4;
            } else {
                float4 a4 = *(const float4*)&Af[(m0 + r) * DMODEL + kb + c];
                ldsA[r][c + 0] = f2bf(a4.x);
                ldsA[r][c + 1] = f2bf(a4.y);
                ldsA[r][c + 2] = f2bf(a4.z);
                ldsA[r][c + 3] = f2bf(a4.w);
            }
        }
        // --- stage W tile 32x128 transposed into [n][k] ---
        #pragma unroll
        for (int p = 0; p < 4; ++p) {
            int idx = p * 1024 + tid * 4;
            int kr = idx >> 7, nc = idx & 127;
            float4 w4 = *(const float4*)&W[(kb + kr) * DMODEL + n0 + nc];
            ldsB[nc + 0][kr] = f2bf(w4.x);
            ldsB[nc + 1][kr] = f2bf(w4.y);
            ldsB[nc + 2][kr] = f2bf(w4.z);
            ldsB[nc + 3][kr] = f2bf(w4.w);
        }
        // --- prefetch next k-tile (gfx1250 global_prefetch_b8) ---
        if (kb + 32 < DMODEL) {
            int idx = tid * 4;
            int r = idx >> 5, c = idx & 31;
            if (A_BF16) __builtin_prefetch(&Ab[(m0 + r) * DMODEL + kb + 32 + c], 0, 1);
            else        __builtin_prefetch(&Af[(m0 + r) * DMODEL + kb + 32 + c], 0, 1);
            int kr = idx >> 7, nc = idx & 127;
            __builtin_prefetch(&W[(kb + 32 + kr) * DMODEL + n0 + nc], 0, 1);
        }
        __syncthreads();

        Frag16 afr[2], bfr[4];
        #pragma unroll
        for (int ti = 0; ti < 2; ++ti) {
            int row = wm * 32 + ti * 16 + (lane & 15);
            afr[ti].h[0] = *(const v8bf*)&ldsA[row][hlf * 8];
            afr[ti].h[1] = *(const v8bf*)&ldsA[row][16 + hlf * 8];
        }
        #pragma unroll
        for (int tj = 0; tj < 4; ++tj) {
            int col = wn * 64 + tj * 16 + (lane & 15);
            bfr[tj].h[0] = *(const v8bf*)&ldsB[col][hlf * 8];
            bfr[tj].h[1] = *(const v8bf*)&ldsB[col][16 + hlf * 8];
        }
        #pragma unroll
        for (int ti = 0; ti < 2; ++ti)
            #pragma unroll
            for (int tj = 0; tj < 4; ++tj)
                acc[ti][tj] = wmma_bf16(afr[ti], bfr[tj], acc[ti][tj]);
        __syncthreads();
    }

    // epilogue: bias + store
    #pragma unroll
    for (int ti = 0; ti < 2; ++ti) {
        #pragma unroll
        for (int tj = 0; tj < 4; ++tj) {
            int nn = n0 + wn * 64 + tj * 16 + (lane & 15);
            float bv = bias[nn];
            #pragma unroll
            for (int r = 0; r < 8; ++r) {
                int mm = m0 + wm * 32 + ti * 16 + r + hlf * 8;
                float val = acc[ti][tj][r] + bv;
                int b = mm >> 11, s = mm & (SEQ - 1);
                int h = nn >> 6,  d = nn & 63;
                if (OUT_MODE == 1) {
                    ((unsigned short*)outPtr)[((((size_t)(b * NHEADS + h) * SEQ) + s) << 6) + d] = f2bf(val);
                } else if (OUT_MODE == 2) {
                    ((unsigned short*)outPtr)[((size_t)((b * NHEADS + h) * DKH + d) << 11) + s] = f2bf(val);
                } else {
                    // write-once fp32 result: non-temporal, keep L2 for tiles
                    __builtin_nontemporal_store(val, &((float*)outPtr)[(size_t)mm * DMODEL + nn]);
                }
            }
        }
    }
}

// ---------------------------------------------------------------------------
// Fused attention: one block = 16 query rows of one (b,h).
// Phase 1: scores QK^T (8 waves x 256-key strips, 16 acc tiles each).
// Phase 2: two-level softmax (shuffle over columns, LDS over waves);
//          writes fp32 probabilities to the attn region of d_out with
//          NON-TEMPORAL stores (512 MB write-once stream must not evict the
//          L2-resident K/V heads reused by 128 blocks each).
// Phase 3: P@V without re-reading P from HBM: each wave round-trips one
//          16x32 bf16 P tile-pair through wave-private LDS (DS in-order,
//          no barrier) to re-layout C->A fragments, B-frags read from the
//          transposed V [B,H,64,S] as contiguous 16B global loads.
//          Deterministic fixed-order cross-wave reduction in LDS, then
//          bf16 attn_out written in [B,S,1024] layout for the final GEMM.
// ---------------------------------------------------------------------------
__global__ __launch_bounds__(256) void attn_fused_k(const unsigned short* __restrict__ Qh,
                                                    const unsigned short* __restrict__ Kh,
                                                    const unsigned short* __restrict__ Vt,
                                                    float* __restrict__ attn,
                                                    unsigned short* __restrict__ AO) {
    __shared__ __align__(16) unsigned short ldsQ[16][72];
    __shared__ __align__(16) unsigned short Pst[8][16][40];  // wave-private P staging
    __shared__ float redA[8][16];
    __shared__ float redB[8][16];
    __shared__ float rowMax[16];
    __shared__ float rowInv[16];
    __shared__ float Opart[8][16][64];                       // per-wave O partials

    const int tid  = threadIdx.x;
    const int lane = tid & 31;
    const int wave = tid >> 5;
    const int hlf  = lane >> 4;
    const int bh   = blockIdx.x >> 7;          // b*16 + h
    const int q0   = (blockIdx.x & 127) << 4;  // query tile base
    const int b    = bh >> 4, h = bh & 15;

    const unsigned short* Qp  = Qh + ((size_t)bh * SEQ + q0) * DKH;
    const unsigned short* Kp  = Kh + (size_t)bh * SEQ * DKH;
    const unsigned short* VtP = Vt + (size_t)bh * DKH * SEQ;
    float* attnP = attn + (size_t)bh * SEQ * SEQ + (size_t)q0 * SEQ;

    {   // stage Q tile 16x64 bf16
        int idx = tid * 4;
        int r = idx >> 6, c = idx & 63;
        ushort4 q4 = *(const ushort4*)&Qp[r * DKH + c];
        *(ushort4*)&ldsQ[r][c] = q4;
    }
    __syncthreads();

    Frag16 aQ[2];
    {
        int row = lane & 15;
        #pragma unroll
        for (int kh = 0; kh < 2; ++kh) {
            aQ[kh].h[0] = *(const v8bf*)&ldsQ[row][kh * 32 + hlf * 8];
            aQ[kh].h[1] = *(const v8bf*)&ldsQ[row][kh * 32 + 16 + hlf * 8];
        }
    }

    const v8f vzero = {};
    v8f acc[16];
    #pragma unroll
    for (int t = 0; t < 16; ++t) acc[t] = vzero;

    const int keyw = wave * 256;
    #pragma unroll
    for (int t = 0; t < 16; ++t) {
        // B-fragment of K^T: lane = key column; per-head K rows contiguous in d
        const unsigned short* kp = Kp + (size_t)(keyw + t * 16 + (lane & 15)) * DKH;
        #pragma unroll
        for (int kh = 0; kh < 2; ++kh) {
            Frag16 bK;
            int base = kh * 32 + hlf * 8;
            bK.h[0] = *(const v8bf*)&kp[base];
            bK.h[1] = *(const v8bf*)&kp[base + 16];
            acc[t] = wmma_bf16(aQ[kh], bK, acc[t]);
        }
    }

    // scale by 1/sqrt(d_k) = 0.125
    #pragma unroll
    for (int t = 0; t < 16; ++t)
        #pragma unroll
        for (int r = 0; r < 8; ++r) acc[t][r] *= 0.125f;

    // row max: per-lane over tiles, shuffle over 16 columns, LDS over waves
    float lred[8];
    #pragma unroll
    for (int r = 0; r < 8; ++r) {
        float m = acc[0][r];
        #pragma unroll
        for (int t = 1; t < 16; ++t) m = fmaxf(m, acc[t][r]);
        m = fmaxf(m, __shfl_xor(m, 1));
        m = fmaxf(m, __shfl_xor(m, 2));
        m = fmaxf(m, __shfl_xor(m, 4));
        m = fmaxf(m, __shfl_xor(m, 8));
        lred[r] = m;
    }
    if ((lane & 15) == 0) {
        #pragma unroll
        for (int r = 0; r < 8; ++r) redA[wave][r + hlf * 8] = lred[r];
    }
    __syncthreads();
    if (tid < 16) {
        float m = redA[0][tid];
        for (int w = 1; w < 8; ++w) m = fmaxf(m, redA[w][tid]);
        rowMax[tid] = m;
    }
    __syncthreads();

    // exp + row sum
    #pragma unroll
    for (int r = 0; r < 8; ++r) {
        float rm = rowMax[r + hlf * 8];
        float s = 0.f;
        #pragma unroll
        for (int t = 0; t < 16; ++t) {
            float e = __expf(acc[t][r] - rm);
            acc[t][r] = e;
            s += e;
        }
        s += __shfl_xor(s, 1);
        s += __shfl_xor(s, 2);
        s += __shfl_xor(s, 4);
        s += __shfl_xor(s, 8);
        lred[r] = s;
    }
    if ((lane & 15) == 0) {
        #pragma unroll
        for (int r = 0; r < 8; ++r) redB[wave][r + hlf * 8] = lred[r];
    }
    __syncthreads();
    if (tid < 16) {
        float s = 0.f;
        for (int w = 0; w < 8; ++w) s += redB[w][tid];
        rowInv[tid] = 1.0f / s;
    }
    __syncthreads();

    // normalize (keep probs in registers) + write fp32 attn (non-temporal)
    #pragma unroll
    for (int r = 0; r < 8; ++r) {
        float inv = rowInv[r + hlf * 8];
        size_t rowoff = (size_t)(r + hlf * 8) * SEQ;
        #pragma unroll
        for (int t = 0; t < 16; ++t) {
            float p = acc[t][r] * inv;
            acc[t][r] = p;
            __builtin_nontemporal_store(p, &attnP[rowoff + keyw + t * 16 + (lane & 15)]);
        }
    }

    // --- Phase 3: O_w = P_w(16x256) @ V_w(256x64), P never leaves the WGP ---
    v8f oacc[4];
    #pragma unroll
    for (int j = 0; j < 4; ++j) oacc[j] = vzero;

    #pragma unroll
    for (int kstep = 0; kstep < 8; ++kstep) {
        // C-layout -> LDS (wave-private strip): one 16x32 bf16 tile-pair
        #pragma unroll
        for (int tt = 0; tt < 2; ++tt) {
            int t = kstep * 2 + tt;
            #pragma unroll
            for (int r = 0; r < 8; ++r)
                Pst[wave][r + hlf * 8][tt * 16 + (lane & 15)] = f2bf(acc[t][r]);
        }
        // A-fragment readback (same-wave DS ops are in-order: no barrier)
        Frag16 aP;
        {
            int row = lane & 15;
            aP.h[0] = *(const v8bf*)&Pst[wave][row][hlf * 8];
            aP.h[1] = *(const v8bf*)&Pst[wave][row][16 + hlf * 8];
        }
        #pragma unroll
        for (int tj = 0; tj < 4; ++tj) {
            Frag16 bV;
            const unsigned short* vp = VtP + (size_t)(tj * 16 + (lane & 15)) * SEQ
                                       + keyw + kstep * 32 + hlf * 8;
            bV.h[0] = *(const v8bf*)&vp[0];
            bV.h[1] = *(const v8bf*)&vp[16];
            oacc[tj] = wmma_bf16(aP, bV, oacc[tj]);
        }
    }

    // deterministic fixed-order cross-wave reduction
    #pragma unroll
    for (int tj = 0; tj < 4; ++tj)
        #pragma unroll
        for (int r = 0; r < 8; ++r)
            Opart[wave][r + hlf * 8][tj * 16 + (lane & 15)] = oacc[tj][r];
    __syncthreads();
    {
        int r  = tid >> 4;
        int c0 = (tid & 15) * 4;
        float s0 = 0.f, s1 = 0.f, s2 = 0.f, s3 = 0.f;
        #pragma unroll
        for (int w = 0; w < 8; ++w) {
            s0 += Opart[w][r][c0 + 0];
            s1 += Opart[w][r][c0 + 1];
            s2 += Opart[w][r][c0 + 2];
            s3 += Opart[w][r][c0 + 3];
        }
        ushort4 o4;
        o4.x = f2bf(s0); o4.y = f2bf(s1); o4.z = f2bf(s2); o4.w = f2bf(s3);
        *(ushort4*)&AO[(size_t)(b * SEQ + q0 + r) * DMODEL + h * DKH + c0] = o4;
    }
}

// ---------------------------------------------------------------------------
extern "C" void kernel_launch(void* const* d_in, const int* in_sizes, int n_in,
                              void* d_out, int out_size, void* d_ws, size_t ws_size,
                              hipStream_t stream) {
    (void)in_sizes; (void)n_in; (void)out_size; (void)ws_size;

    const float* query = (const float*)d_in[0];
    const float* key   = (const float*)d_in[1];
    const float* value = (const float*)d_in[2];
    const float* Wq = (const float*)d_in[3];
    const float* bq = (const float*)d_in[4];
    const float* Wk = (const float*)d_in[5];
    const float* bk = (const float*)d_in[6];
    const float* Wv = (const float*)d_in[7];
    const float* bv = (const float*)d_in[8];
    const float* Wo = (const float*)d_in[9];
    const float* bo = (const float*)d_in[10];

    float* outO = (float*)d_out;                                // [B,S,1024] fp32
    float* attn = (float*)d_out + (size_t)BATCH * SEQ * DMODEL; // [B,H,S,S] fp32

    // workspace: Qh | Kh ([B,H,S,64] bf16) | Vt ([B,H,64,S] bf16) | AO ([B,S,1024] bf16)
    const size_t headElems = (size_t)BATCH * NHEADS * SEQ * DKH;   // 4,194,304
    unsigned short* Qh = (unsigned short*)d_ws;
    unsigned short* Kh = Qh + headElems;
    unsigned short* Vt = Kh + headElems;
    unsigned short* AO = Vt + headElems;

    dim3 blk(256);
    const int gemmGrid = (MROWS / 128) * (DMODEL / 128);   // 256
    const int attnGrid = BATCH * NHEADS * (SEQ / 16);      // 4096

    gemm_bias_k<false, 1><<<gemmGrid, blk, 0, stream>>>(query, Wq, bq, Qh);
    gemm_bias_k<false, 1><<<gemmGrid, blk, 0, stream>>>(key,   Wk, bk, Kh);
    gemm_bias_k<false, 2><<<gemmGrid, blk, 0, stream>>>(value, Wv, bv, Vt);
    attn_fused_k<<<attnGrid, blk, 0, stream>>>(Qh, Kh, Vt, attn, AO);
    gemm_bias_k<true, 0><<<gemmGrid, blk, 0, stream>>>(AO, Wo, bo, outO);
}
